// GLiNERLinkerPooler_14671608283278
// MI455X (gfx1250) — compile-verified, matrix operand order checked by reading, not run on hardware
//
#include <hip/hip_runtime.h>
#include <hip/hip_bf16.h>

// ---------------------------------------------------------------------------
// GLiNER linker/pooler, factored:
//   we = scatter(tok_embs, wmask)                     (1024 x 768)
//   t0/t1 = we @ W_tok halves + b_tok halves          (bf16 WMMA GEMM)
//   l0/l1 = label_embs @ W_lab halves + b_lab halves
//   U = t0 @ W1a + b1   (per-word),  V = l0 @ W1b     (per-label)
//   scores[w,c,:] = relu( (t1[w]*l1[c]) @ W1c + U[w] + V[c] ) @ W2 + b2
// Main kernel: one block per word, all 64 labels; l1 async-staged to LDS
// (global_load_async_to_lds_b128), p = t1*l1 in place; each wave reuses one
// W1c^T fragment across 4 M-subtiles (4 WMMAs / B-load); LDS cross-wave
// reduction, no atomics.
// ---------------------------------------------------------------------------

#define H 768
#define LTOK 2048
#define WN 1024
#define CN 64
#define PPITCH 776   // padded LDS row pitch (bf16 elems): 1552B, 16B aligned

typedef __attribute__((ext_vector_type(16))) __bf16 v16bf;
typedef __attribute__((ext_vector_type(8)))  float  v8f;

union Frag {
    uint4 u[2];
    v16bf v;
};

__device__ __forceinline__ float bf2f(unsigned short u) {
    unsigned int x = ((unsigned int)u) << 16;
    return __builtin_bit_cast(float, x);
}
__device__ __forceinline__ unsigned short f2bf(float f) {
    unsigned int x = __builtin_bit_cast(unsigned int, f);
    unsigned int r = x + 0x7FFFu + ((x >> 16) & 1u);
    return (unsigned short)(r >> 16);
}

// ---- init: zero `we` ------------------------------------------------------
__global__ void k_init(float* __restrict__ we) {
    int i = blockIdx.x * 256 + threadIdx.x;
    if (i < WN * H) we[i] = 0.0f;
}

// ---- scatter pooling: we[wmask-1] = tok_embs ------------------------------
__global__ void k_scatter(const float* __restrict__ tok,
                          const int* __restrict__ wmask,
                          float* __restrict__ we) {
    int i = blockIdx.x * 256 + threadIdx.x;       // over L*H
    int l = i / H, h = i - l * H;
    int w = wmask[l];
    if (w > 0) we[(size_t)(w - 1) * H + h] = tok[i];
}

// ---- transpose + f32->bf16: dst[n*768+k] = src[k*ld + n], 768x768 ---------
__global__ void k_transconv(const float* __restrict__ src, int ld,
                            unsigned short* __restrict__ dst) {
    int i = blockIdx.x * 256 + threadIdx.x;       // over 768*768
    int n = i / H, k = i - n * H;
    dst[(size_t)n * H + k] = f2bf(src[(size_t)k * ld + n]);
}

// ---- f32 -> bf16 copy -----------------------------------------------------
__global__ void k_conv(const float* __restrict__ src,
                       unsigned short* __restrict__ dst, int nelem) {
    int i = blockIdx.x * 256 + threadIdx.x;
    if (i < nelem) dst[i] = f2bf(src[i]);
}

// ---- generic bf16 WMMA GEMM: D(MxN,f32) = A(MxK,bf16) @ B^T(NxK,bf16)+bias
__global__ void __launch_bounds__(128)
k_gemm(const unsigned short* __restrict__ A,
       const unsigned short* __restrict__ BT,
       const float* __restrict__ bias,
       float* __restrict__ D, int M, int N, int K) {
    int wave = threadIdx.x >> 5;
    int lane = threadIdx.x & 31;
    int tile = blockIdx.x * 4 + wave;
    int ntiles = N >> 4;
    int mt = tile / ntiles, nt = tile - mt * ntiles;
    int m0 = mt << 4, n0 = nt << 4;
    int lm = lane & 15, sel = lane >> 4;

    const unsigned short* arow = A  + (size_t)(m0 + lm) * K;
    const unsigned short* brow = BT + (size_t)(n0 + lm) * K;

    v8f acc = {};
#pragma unroll 4
    for (int kb = 0; kb < K; kb += 32) {
        Frag a, b;
        a.u[0] = *(const uint4*)(arow + kb + sel * 8);
        a.u[1] = *(const uint4*)(arow + kb + 16 + sel * 8);
        b.u[0] = *(const uint4*)(brow + kb + sel * 8);
        b.u[1] = *(const uint4*)(brow + kb + 16 + sel * 8);
        acc = __builtin_amdgcn_wmma_f32_16x16x32_bf16(
            false, a.v, false, b.v, (short)0, acc, false, false);
    }
    float bv = bias ? bias[n0 + lm] : 0.0f;
#pragma unroll
    for (int r = 0; r < 8; ++r)
        D[(size_t)(m0 + r + (sel << 3)) * N + n0 + lm] = acc[r] + bv;
}

// ---- fused main: block = (w, all 64 labels) -------------------------------
__global__ void __launch_bounds__(256)
k_main(const unsigned short* __restrict__ t1b,
       const unsigned short* __restrict__ l1b,
       const unsigned short* __restrict__ W1cT,
       const float* __restrict__ U, const float* __restrict__ V,
       const float* __restrict__ W2, const float* __restrict__ b2,
       float* __restrict__ out) {
    __shared__ __align__(16) unsigned short p[CN * PPITCH];   // 99328 B

    int w = blockIdx.x;
    int tid = threadIdx.x;

    // Phase 1: async-stage l1 (64x768 bf16) into padded LDS rows
    {
        unsigned int base_lds = (unsigned int)(size_t)(&p[0]);
        unsigned long long gbase = (unsigned long long)(size_t)l1b;
#pragma unroll 4
        for (int j = 0; j < 24; ++j) {
            int li = j * 256 + tid;           // 0..6143 (vec8 index)
            int c  = li / 96;                 // 768/8 = 96 vec8 per row
            int k  = (li - c * 96) * 8;
            unsigned int lds = base_lds + (unsigned int)(c * PPITCH + k) * 2u;
            unsigned long long ga = gbase + (unsigned long long)(c * H + k) * 2u;
            asm volatile("global_load_async_to_lds_b128 %0, %1, off"
                         :: "v"(lds), "v"(ga) : "memory");
        }
        asm volatile("s_wait_asynccnt 0x0" ::: "memory");
    }
    __syncthreads();

    // Phase 2: p[c][k] *= t1[w][k]   (in place, bf16)
    {
        int c  = tid >> 2;
        int k0 = (tid & 3) * 192;
        const unsigned short* tr = t1b + (size_t)w * H + k0;
        unsigned short* pr = p + c * PPITCH + k0;
#pragma unroll 8
        for (int jj = 0; jj < 192; ++jj)
            pr[jj] = f2bf(bf2f(pr[jj]) * bf2f(tr[jj]));
    }
    __syncthreads();

    // Phase 3: WMMA; wave handles N-tiles nt = wave + 8*i; B frag reused for
    // 4 M-subtiles (c-quarters)
    int wave = tid >> 5, lane = tid & 31;
    int lm = lane & 15, sel = lane >> 4;

    float s3[4][8][3];
#pragma unroll
    for (int ms = 0; ms < 4; ++ms)
#pragma unroll
        for (int r = 0; r < 8; ++r)
            s3[ms][r][0] = s3[ms][r][1] = s3[ms][r][2] = 0.0f;

    const unsigned short* pr0 = p + (0  + lm) * PPITCH;
    const unsigned short* pr1 = p + (16 + lm) * PPITCH;
    const unsigned short* pr2 = p + (32 + lm) * PPITCH;
    const unsigned short* pr3 = p + (48 + lm) * PPITCH;

    for (int i = 0; i < 6; ++i) {
        int nt = wave + (i << 3);            // 0..47
        int n0 = nt << 4;
        const unsigned short* brow = W1cT + (size_t)(n0 + lm) * H;
        if (i < 5)
            __builtin_prefetch(W1cT + (size_t)(n0 + 128 + lm) * H, 0, 1);

        v8f acc[4];
#pragma unroll
        for (int ms = 0; ms < 4; ++ms) acc[ms] = (v8f){};

#pragma unroll 2
        for (int kb = 0; kb < H; kb += 32) {
            int o0 = kb + sel * 8, o1 = kb + 16 + sel * 8;
            Frag bfr;
            bfr.u[0] = *(const uint4*)(brow + o0);
            bfr.u[1] = *(const uint4*)(brow + o1);
            Frag fa;
            fa.u[0] = *(const uint4*)(pr0 + o0);
            fa.u[1] = *(const uint4*)(pr0 + o1);
            acc[0] = __builtin_amdgcn_wmma_f32_16x16x32_bf16(
                false, fa.v, false, bfr.v, (short)0, acc[0], false, false);
            fa.u[0] = *(const uint4*)(pr1 + o0);
            fa.u[1] = *(const uint4*)(pr1 + o1);
            acc[1] = __builtin_amdgcn_wmma_f32_16x16x32_bf16(
                false, fa.v, false, bfr.v, (short)0, acc[1], false, false);
            fa.u[0] = *(const uint4*)(pr2 + o0);
            fa.u[1] = *(const uint4*)(pr2 + o1);
            acc[2] = __builtin_amdgcn_wmma_f32_16x16x32_bf16(
                false, fa.v, false, bfr.v, (short)0, acc[2], false, false);
            fa.u[0] = *(const uint4*)(pr3 + o0);
            fa.u[1] = *(const uint4*)(pr3 + o1);
            acc[3] = __builtin_amdgcn_wmma_f32_16x16x32_bf16(
                false, fa.v, false, bfr.v, (short)0, acc[3], false, false);
        }

        int n = n0 + lm;
        float un  = U[(size_t)w * H + n];
        float w20 = W2[n * 3 + 0], w21 = W2[n * 3 + 1], w22 = W2[n * 3 + 2];
#pragma unroll
        for (int ms = 0; ms < 4; ++ms)
#pragma unroll
            for (int r = 0; r < 8; ++r) {
                int c = ms * 16 + r + (sel << 3);
                float h = acc[ms][r] + un + V[(size_t)c * H + n];
                h = fmaxf(h, 0.0f);
                s3[ms][r][0] += h * w20;
                s3[ms][r][1] += h * w21;
                s3[ms][r][2] += h * w22;
            }
    }

    // reduce over the 16 lanes that hold different n
#pragma unroll
    for (int off = 8; off >= 1; off >>= 1)
#pragma unroll
        for (int ms = 0; ms < 4; ++ms)
#pragma unroll
            for (int r = 0; r < 8; ++r) {
                s3[ms][r][0] += __shfl_xor(s3[ms][r][0], off, 16);
                s3[ms][r][1] += __shfl_xor(s3[ms][r][1], off, 16);
                s3[ms][r][2] += __shfl_xor(s3[ms][r][2], off, 16);
            }

    // cross-wave reduction via LDS (reuse p), then plain stores (no atomics)
    __syncthreads();
    float* part = (float*)p;          // [8 waves][64 pairs][3]
    if (lm == 0) {
#pragma unroll
        for (int ms = 0; ms < 4; ++ms)
#pragma unroll
            for (int r = 0; r < 8; ++r) {
                int c = ms * 16 + r + (sel << 3);
                float* dst = part + wave * (CN * 3) + c * 3;
                dst[0] = s3[ms][r][0];
                dst[1] = s3[ms][r][1];
                dst[2] = s3[ms][r][2];
            }
    }
    __syncthreads();
    if (tid < CN * 3) {
        float s = 0.0f;
#pragma unroll
        for (int wv = 0; wv < 8; ++wv) s += part[wv * (CN * 3) + tid];
        out[(size_t)w * (CN * 3) + tid] = b2[tid % 3] + s;
    }
}

// ---------------------------------------------------------------------------
extern "C" void kernel_launch(void* const* d_in, const int* in_sizes, int n_in,
                              void* d_out, int out_size, void* d_ws, size_t ws_size,
                              hipStream_t stream) {
    const float* tok   = (const float*)d_in[0];
    const int*   wmask = (const int*)d_in[1];
    // d_in[2] text_length == 1024 (compile-time constant here)
    const float* lab   = (const float*)d_in[3];
    const float* W_tok = (const float*)d_in[4];
    const float* b_tok = (const float*)d_in[5];
    const float* W_lab = (const float*)d_in[6];
    const float* b_lab = (const float*)d_in[7];
    const float* W1    = (const float*)d_in[8];
    const float* b1    = (const float*)d_in[9];
    const float* W2    = (const float*)d_in[10];
    const float* b2    = (const float*)d_in[11];
    float* out = (float*)d_out;

    char* ws = (char*)d_ws;
    size_t off = 0;
    auto alloc = [&](size_t bytes) -> void* {
        void* p = ws + off;
        off += (bytes + 255) & ~(size_t)255;
        return p;
    };

    float* we   = (float*)alloc((size_t)WN * H * 4);
    float* t0   = (float*)alloc((size_t)WN * H * 4);
    float* t1   = (float*)alloc((size_t)WN * H * 4);
    float* Ubuf = (float*)alloc((size_t)WN * H * 4);
    float* l0   = (float*)alloc((size_t)CN * H * 4);
    float* l1   = (float*)alloc((size_t)CN * H * 4);
    float* Vbuf = (float*)alloc((size_t)CN * H * 4);

    unsigned short* we_b  = (unsigned short*)alloc((size_t)WN * H * 2);
    unsigned short* lab_b = (unsigned short*)alloc((size_t)CN * H * 2);
    unsigned short* t0b   = (unsigned short*)alloc((size_t)WN * H * 2);
    unsigned short* t1b   = (unsigned short*)alloc((size_t)WN * H * 2);
    unsigned short* l0b   = (unsigned short*)alloc((size_t)CN * H * 2);
    unsigned short* l1b   = (unsigned short*)alloc((size_t)CN * H * 2);

    unsigned short* WtokT0 = (unsigned short*)alloc((size_t)H * H * 2);
    unsigned short* WtokT1 = (unsigned short*)alloc((size_t)H * H * 2);
    unsigned short* WlabT0 = (unsigned short*)alloc((size_t)H * H * 2);
    unsigned short* WlabT1 = (unsigned short*)alloc((size_t)H * H * 2);
    unsigned short* W1aT   = (unsigned short*)alloc((size_t)H * H * 2);
    unsigned short* W1bT   = (unsigned short*)alloc((size_t)H * H * 2);
    unsigned short* W1cT   = (unsigned short*)alloc((size_t)H * H * 2);

    // 1) init we=0
    k_init<<<(WN * H + 255) / 256, 256, 0, stream>>>(we);
    // 2) scatter pooling
    k_scatter<<<(LTOK * H) / 256, 256, 0, stream>>>(tok, wmask, we);
    // 3) weight transposes to bf16
    const int TC = (H * H) / 256;
    k_transconv<<<TC, 256, 0, stream>>>(W_tok,          2 * H, WtokT0);
    k_transconv<<<TC, 256, 0, stream>>>(W_tok + H,      2 * H, WtokT1);
    k_transconv<<<TC, 256, 0, stream>>>(W_lab,          2 * H, WlabT0);
    k_transconv<<<TC, 256, 0, stream>>>(W_lab + H,      2 * H, WlabT1);
    k_transconv<<<TC, 256, 0, stream>>>(W1,                 H, W1aT);
    k_transconv<<<TC, 256, 0, stream>>>(W1 + (size_t)H * H,     H, W1bT);
    k_transconv<<<TC, 256, 0, stream>>>(W1 + (size_t)2 * H * H, H, W1cT);
    // 4) activation converts
    k_conv<<<(WN * H + 255) / 256, 256, 0, stream>>>(we, we_b, WN * H);
    k_conv<<<(CN * H + 255) / 256, 256, 0, stream>>>(lab, lab_b, CN * H);
    // 5) scorer projections (bf16 WMMA)
    const int GB_W = (WN / 16) * (H / 16) / 4;  // 768 blocks
    const int GB_C = (CN / 16) * (H / 16) / 4;  // 48 blocks
    k_gemm<<<GB_W, 128, 0, stream>>>(we_b,  WtokT0, b_tok,     t0, WN, H, H);
    k_gemm<<<GB_W, 128, 0, stream>>>(we_b,  WtokT1, b_tok + H, t1, WN, H, H);
    k_gemm<<<GB_C, 128, 0, stream>>>(lab_b, WlabT0, b_lab,     l0, CN, H, H);
    k_gemm<<<GB_C, 128, 0, stream>>>(lab_b, WlabT1, b_lab + H, l1, CN, H, H);
    // 6) convert projections to bf16
    k_conv<<<(WN * H + 255) / 256, 256, 0, stream>>>(t0, t0b, WN * H);
    k_conv<<<(WN * H + 255) / 256, 256, 0, stream>>>(t1, t1b, WN * H);
    k_conv<<<(CN * H + 255) / 256, 256, 0, stream>>>(l0, l0b, CN * H);
    k_conv<<<(CN * H + 255) / 256, 256, 0, stream>>>(l1, l1b, CN * H);
    // 7) per-w and per-c partial MLP terms
    k_gemm<<<GB_W, 128, 0, stream>>>(t0b, W1aT, b1,      Ubuf, WN, H, H);
    k_gemm<<<GB_C, 128, 0, stream>>>(l0b, W1bT, nullptr, Vbuf, CN, H, H);
    // 8) fused pairwise GEMM + relu + W2 contraction (one block per word)
    k_main<<<WN, 256, 0, stream>>>(t1b, l1b, W1cT, Ubuf, Vbuf, W2, b2, out);
}
